// GWM_10247791968408
// MI455X (gfx1250) — matrix operand order repaired and checked by reading, not compiled, not run on hardware
//
#include <hip/hip_runtime.h>
#include <hip/hip_bf16.h>

typedef __attribute__((ext_vector_type(16))) _Float16 v16h;
typedef __attribute__((ext_vector_type(8)))  _Float16 v8h;
typedef __attribute__((ext_vector_type(8)))  float    v8f;
typedef __attribute__((ext_vector_type(4)))  float    v4f;

#define B_    4
#define N_    4096
#define C_    256
#define NH    8
#define NG    4
#define HD    32
#define NLOC  1024                    // N_/NG
#define HGN   (B_*NG*NH)              // 128 head-groups
#define SCALE 0.17677669529663687f    // 32^-0.5

// ---- WMMA fragment loaders (layouts per CDNA5 ISA §7.12.2) ----
// A (16x32 f16): lane holds row M=lane%16; elems 0..7 = K c0+0..7, elems 8..15 = K c0+16..23,
// c0 = 8*(lane/16). Caller passes pointer to (row, c0).
static __device__ __forceinline__ v16h a_frag_f16(const _Float16* p) {
  v8h lo = *(const v8h*)(p);
  v8h hi = *(const v8h*)(p + 16);
  v16h a;
#pragma unroll
  for (int i = 0; i < 8; ++i) { a[i] = lo[i]; a[i + 8] = hi[i]; }
  return a;
}
static __device__ __forceinline__ v16h a_frag_f32(const float* p) {
  v4f x0 = *(const v4f*)(p);
  v4f x1 = *(const v4f*)(p + 4);
  v4f x2 = *(const v4f*)(p + 16);
  v4f x3 = *(const v4f*)(p + 20);
  v16h a;
#pragma unroll
  for (int i = 0; i < 4; ++i) {
    a[i]      = (_Float16)x0[i];
    a[i + 4]  = (_Float16)x1[i];
    a[i + 8]  = (_Float16)x2[i];
    a[i + 12] = (_Float16)x3[i];
  }
  return a;
}
// B (32x16 f16): lane holds column N=lane%16; elems 0..7 = K kb+0..7, elems 8..15 = K kb+16..23, kb = 8*(lane/16).
// Source must be column-major in B == row-major in B^T; caller passes (col, kb) pointer.
static __device__ __forceinline__ v16h b_frag_f16(const _Float16* p) {
  v8h lo = *(const v8h*)(p);
  v8h hi = *(const v8h*)(p + 16);
  v16h b;
#pragma unroll
  for (int i = 0; i < 8; ++i) { b[i] = lo[i]; b[i + 8] = hi[i]; }
  return b;
}
static __device__ __forceinline__ v8f wmma_f16(v16h a, v16h b, v8f c) {
  return __builtin_amdgcn_wmma_f32_16x16x32_f16(false, a, false, b, (short)0, c,
                                                false, false);
}

// -------- Kernel 0: one-shot f32 -> f16 weight conversion (L2-resident) ----
__global__ __launch_bounds__(256) void cvt_kernel(
    const float* __restrict__ wq, const float* __restrict__ wp,
    _Float16* __restrict__ wq16, _Float16* __restrict__ wp16) {
  const int t = blockIdx.x * blockDim.x + threadIdx.x;   // 65536 threads
#pragma unroll
  for (int i = 0; i < 3; ++i)                            // 768*256 = 3*65536
    wq16[t + i * 65536] = (_Float16)wq[t + i * 65536];
  wp16[t] = (_Float16)wp[t];                             // 256*256 = 65536
}

// ------------------- Kernel 1: fused permute + QKV GEMM -------------------
// out(i, j) = x[b, idx[i], :] . w_qkv[j, :]   M=B*N, N=768, K=256
// Each wave computes a 16x64 strip (A-frag reused across 4 WMMAs per K-step).
// q/k scattered to [hg][n][d] f16, v scattered transposed to [hg][d][n] f16.
__global__ __launch_bounds__(256) void qkv_kernel(
    const float* __restrict__ x, const int* __restrict__ idx,
    const _Float16* __restrict__ wq16, _Float16* __restrict__ qb,
    _Float16* __restrict__ kb, _Float16* __restrict__ vb) {
  const int wave = blockIdx.x * (blockDim.x >> 5) + (threadIdx.x >> 5);
  const int mt = wave / 12;    // [0,1024) : 16-row tiles over B*N (permuted order)
  const int ntq = wave % 12;   // [0,12)   : 64-col strips over 3C=768
  const int lane = threadIdx.x & 31;
  const int l16 = lane & 15;
  const int lh = lane >> 4;

  const int b = mt >> 8;                       // 256 row-tiles per batch
  const int irow = ((mt & 255) << 4) + l16;    // permuted position (A row)
  const int src = idx[irow];                   // gather source row
  const float* ap = x + ((size_t)b * N_ + src) * C_ + lh * 8;
  const _Float16* bp = wq16 + (size_t)(ntq * 64 + l16) * C_ + lh * 8;

  v8f accs[4] = {{}, {}, {}, {}};
#pragma unroll
  for (int kk = 0; kk < 16; ++kk) {            // 16 steps of 32 = 512 k (rows are C_ = 256)
    const v16h a = a_frag_f32(ap + kk * 32);   // loaded once, used 4x
#pragma unroll
    for (int t = 0; t < 4; ++t)
      accs[t] = wmma_f16(a, b_frag_f16(bp + (size_t)t * 16 * C_ + kk * 32), accs[t]);
  }

  // Scatter C tiles: lane holds (M = r + 8*lh, N = l16) per tile t
  const int rowbase = (mt & 255) << 4;
#pragma unroll
  for (int t = 0; t < 4; ++t) {
    const int j = ntq * 64 + t * 16 + l16;     // [0,768): [s=3][h=8][d=32]
    const int s = j >> 8;                      // uniform across wave
    const int h = (j >> 5) & 7;
    const int dd = j & 31;
#pragma unroll
    for (int r = 0; r < 8; ++r) {
      const int row = rowbase + r + lh * 8;    // permuted position within batch
      const int g = row >> 10;
      const int nl = row & 1023;
      const size_t hg = (size_t)(b * NG + g) * NH + h;
      const _Float16 val = (_Float16)accs[t][r];
      if (s == 0)
        qb[hg * NLOC * HD + (size_t)nl * HD + dd] = val;
      else if (s == 1)
        kb[hg * NLOC * HD + (size_t)nl * HD + dd] = val;
      else
        vb[hg * HD * NLOC + (size_t)dd * NLOC + nl] = val;   // transposed
    }
  }
}

// ---------------- Kernel 2: flash attention per head-group ----------------
// 1 block per head-group, 8 waves; each wave owns 16 query rows at a time,
// streams 32-key tiles with online softmax. Output un-permuted via idx.
__global__ __launch_bounds__(256) void attn_kernel(
    const _Float16* __restrict__ qb, const _Float16* __restrict__ kb,
    const _Float16* __restrict__ vb, const int* __restrict__ idx,
    _Float16* __restrict__ ao) {
  __shared__ _Float16 pscr[8][16 * 32];   // per-wave P-tile staging (1 KB each)

  const int hg = blockIdx.x;              // [0,128)
  const int b = hg >> 5;
  const int g = (hg >> 3) & 3;
  const int h = hg & 7;
  const int wid = threadIdx.x >> 5;
  const int lane = threadIdx.x & 31;
  const int l16 = lane & 15;
  const int lh = lane >> 4;

  const _Float16* q = qb + (size_t)hg * NLOC * HD;
  const _Float16* k = kb + (size_t)hg * NLOC * HD;
  const _Float16* vt = vb + (size_t)hg * HD * NLOC;
  _Float16* ps = &pscr[wid][0];

  for (int qt = wid; qt < NLOC / 16; qt += 8) {
    const int qrow = qt * 16 + l16;
    const v16h qa = a_frag_f16(q + (size_t)qrow * HD + lh * 8);  // K = d = 32
    v8f o0 = {}, o1 = {};
    float m[8], l[8];
#pragma unroll
    for (int r = 0; r < 8; ++r) { m[r] = -3.0e38f; l[r] = 0.0f; }

    for (int kt = 0; kt < NLOC / 32; ++kt) {
      const int kbase = kt * 32;
      // prefetch next key/value tile into WGP$ (global_prefetch_b8)
      if (kt + 1 < NLOC / 32) {
        __builtin_prefetch(k + (size_t)(kbase + 32 + l16) * HD + lh * 16, 0, 1);
        __builtin_prefetch(vt + (size_t)l16 * NLOC + kbase + 32 + lh * 16, 0, 1);
      }
      // S = Q K^T : B column = key row, K = d (contiguous in k row)
      v16h kf0 = b_frag_f16(k + (size_t)(kbase + l16) * HD + lh * 8);
      v16h kf1 = b_frag_f16(k + (size_t)(kbase + 16 + l16) * HD + lh * 8);
      v8f z = {};
      v8f s0 = wmma_f16(qa, kf0, z);
      v8f s1 = wmma_f16(qa, kf1, z);
#pragma unroll
      for (int r = 0; r < 8; ++r) {
        const float a0 = s0[r] * SCALE;
        const float a1 = s1[r] * SCALE;
        // row max over 16 lanes of this half-wave (row lives in lanes l16=0..15)
        float rm = fmaxf(a0, a1);
        rm = fmaxf(rm, __shfl_xor(rm, 1));
        rm = fmaxf(rm, __shfl_xor(rm, 2));
        rm = fmaxf(rm, __shfl_xor(rm, 4));
        rm = fmaxf(rm, __shfl_xor(rm, 8));
        const float mn = fmaxf(m[r], rm);
        const float alpha = __expf(m[r] - mn);
        const float p0 = __expf(a0 - mn);
        const float p1 = __expf(a1 - mn);
        float rs = p0 + p1;
        rs += __shfl_xor(rs, 1);
        rs += __shfl_xor(rs, 2);
        rs += __shfl_xor(rs, 4);
        rs += __shfl_xor(rs, 8);
        m[r] = mn;
        l[r] = l[r] * alpha + rs;
        o0[r] *= alpha;
        o1[r] *= alpha;
        // stage P in LDS row-major 16x32 (C-layout -> A-layout relay)
        const int row = r + lh * 8;
        ps[row * 32 + l16] = (_Float16)p0;
        ps[row * 32 + 16 + l16] = (_Float16)p1;
      }
      // wave-internal LDS RAW: DS ops are in-order per wave; wait then reload
      asm volatile("s_wait_dscnt 0" ::: "memory");
      const v16h pa = a_frag_f16(ps + l16 * 32 + lh * 8);        // 16x32, K=keys
      // O += P V : B column = dim, K = key (contiguous in vT row)
      v16h vf0 = b_frag_f16(vt + (size_t)l16 * NLOC + kbase + lh * 8);
      v16h vf1 = b_frag_f16(vt + (size_t)(16 + l16) * NLOC + kbase + lh * 8);
      o0 = wmma_f16(pa, vf0, o0);
      o1 = wmma_f16(pa, vf1, o1);
    }

    // normalize + un-permute scatter: output row i -> position idx[i]
#pragma unroll
    for (int r = 0; r < 8; ++r) {
      const int row = qt * 16 + r + lh * 8;
      const int pos = idx[g * NLOC + row];
      const float inv = 1.0f / l[r];
      _Float16* dst = ao + ((size_t)b * N_ + pos) * C_ + h * HD;
      dst[l16] = (_Float16)(o0[r] * inv);
      dst[16 + l16] = (_Float16)(o1[r] * inv);
    }
  }
}

// ----------------------- Kernel 3: output projection -----------------------
// out = ao @ w_proj^T + b_proj   M=B*N, N=256, K=256, fp32 out
// Each wave computes a 16x64 strip (A-frag reused across 4 WMMAs per K-step).
__global__ __launch_bounds__(256) void proj_kernel(
    const _Float16* __restrict__ ao, const _Float16* __restrict__ wp16,
    const float* __restrict__ b_proj, float* __restrict__ out) {
  const int wave = blockIdx.x * (blockDim.x >> 5) + (threadIdx.x >> 5);
  const int mt = wave >> 2;   // [0,1024)
  const int ntq = wave & 3;   // [0,4) : 64-col strips over C=256
  const int lane = threadIdx.x & 31;
  const int l16 = lane & 15;
  const int lh = lane >> 4;

  const int b = mt >> 8;
  const int rowbase = (mt & 255) << 4;
  const _Float16* ap = ao + ((size_t)b * N_ + rowbase + l16) * C_ + lh * 8;
  const _Float16* bp = wp16 + (size_t)(ntq * 64 + l16) * C_ + lh * 8;

  v8f accs[4] = {{}, {}, {}, {}};
#pragma unroll
  for (int kk = 0; kk < 8; ++kk) {            // K = 256 in steps of 32
    const v16h a = a_frag_f16(ap + kk * 32);  // loaded once, used 4x
#pragma unroll
    for (int t = 0; t < 4; ++t)
      accs[t] = wmma_f16(a, b_frag_f16(bp + (size_t)t * 16 * C_ + kk * 32), accs[t]);
  }

#pragma unroll
  for (int t = 0; t < 4; ++t) {
    const int j = ntq * 64 + t * 16 + l16;
    const float bias = b_proj[j];
#pragma unroll
    for (int r = 0; r < 8; ++r) {
      const int row = rowbase + r + lh * 8;
      out[((size_t)b * N_ + row) * C_ + j] = accs[t][r] + bias;
    }
  }
}

// --------------------------------- launch ---------------------------------
extern "C" void kernel_launch(void* const* d_in, const int* in_sizes, int n_in,
                              void* d_out, int out_size, void* d_ws,
                              size_t ws_size, hipStream_t stream) {
  const float* x = (const float*)d_in[0];
  const int* idx = (const int*)d_in[1];
  const float* w_qkv = (const float*)d_in[2];
  const float* w_proj = (const float*)d_in[3];
  const float* b_proj = (const float*)d_in[4];
  float* out = (float*)d_out;

  // workspace layout (f16): q[128][1024][32], k[128][1024][32],
  // vT[128][32][1024], attn_out[B][N][C], wq16[768][256], wp16[256][256]
  _Float16* qb = (_Float16*)d_ws;
  _Float16* kb = qb + (size_t)HGN * NLOC * HD;
  _Float16* vb = kb + (size_t)HGN * NLOC * HD;
  _Float16* ao = vb + (size_t)HGN * HD * NLOC;
  _Float16* wq16 = ao + (size_t)B_ * N_ * C_;
  _Float16* wp16 = wq16 + (size_t)3 * C_ * C_;

  // one-shot weight conversion (65536 threads)
  cvt_kernel<<<256, 256, 0, stream>>>(w_qkv, w_proj, wq16, wp16);
  // 1024 M-tiles x 12 strip-tiles = 12288 waves, 8 waves/block
  qkv_kernel<<<1536, 256, 0, stream>>>(x, idx, wq16, qb, kb, vb);
  // one block per head-group
  attn_kernel<<<128, 256, 0, stream>>>(qb, kb, vb, idx, ao);
  // 1024 M-tiles x 4 strip-tiles = 4096 waves, 8 waves/block
  proj_kernel<<<512, 256, 0, stream>>>(ao, wp16, b_proj, out);
}
